// T2t_ViM_77300821394096
// MI455X (gfx1250) — matrix operand order repaired and checked by reading, not compiled, not run on hardware
//
#include <hip/hip_runtime.h>
#include <hip/hip_bf16.h>
#include <cstdint>
#include <cstddef>

// T2T Vision-Mamba forward for gfx1250 (MI455X), wave32 + WMMA + async LDS.

typedef _Float16 v16h __attribute__((ext_vector_type(16)));
typedef float    v8f  __attribute__((ext_vector_type(8)));
typedef int      v4i  __attribute__((__vector_size__(16)));
#define AS1 __attribute__((address_space(1)))
#define AS3 __attribute__((address_space(3)))

#if __has_builtin(__builtin_amdgcn_global_load_async_to_lds_b128)
#define HAVE_ASYNC_LDS 1
#endif

// ---------------------------------------------------------------------------
// Cast + pad f32 -> f16:  dst[Rp x Cp] = src[R x C] zero-padded.
// ---------------------------------------------------------------------------
__global__ void k_cast_pad(const float* __restrict__ src,
                           _Float16* __restrict__ dst,
                           int R, int C, int Cp, int total) {
  int idx = blockIdx.x * blockDim.x + threadIdx.x;
  if (idx >= total) return;
  int r = idx / Cp, c = idx % Cp;
  float v = (r < R && c < C) ? src[(size_t)r * C + c] : 0.0f;
  dst[idx] = (_Float16)v;
}

// Cast + pad + TRANSPOSE f32 -> f16: dst[Np x Kp] = src[K x N]^T zero-padded.
__global__ void k_cast_padT(const float* __restrict__ src,
                            _Float16* __restrict__ dst,
                            int K, int N, int Kp, int total) {
  int idx = blockIdx.x * blockDim.x + threadIdx.x;
  if (idx >= total) return;
  int n = idx / Kp, k = idx % Kp;
  float v = (n < N && k < K) ? src[(size_t)k * N + n] : 0.0f;
  dst[idx] = (_Float16)v;
}

// ---------------------------------------------------------------------------
// GEMM: C[M,N] = Ah[Mp,Kp] @ WhT[Np,Kp]^T + bias[N]   (f16 in, f32 out)
// Operands pre-padded & W pre-transposed: NO bounds checks, NO synchronous
// loads in the hot loop — both tiles staged with GLOBAL_LOAD_ASYNC_TO_LDS_B128.
// Block = 128 threads = 4 wave32, each wave one 16x16 D tile (64x16 / block).
// ---------------------------------------------------------------------------
__global__ void __launch_bounds__(128)
k_gemm_f16(const _Float16* __restrict__ Ah, const _Float16* __restrict__ WhT,
           const float* __restrict__ bias, float* __restrict__ C,
           int M, int N, int Kp) {
  __shared__ _Float16 Alds[64][32];   // row-major A tile (64B rows)
  __shared__ _Float16 Blds[16][32];   // Blds[n][k] = 16 rows of WhT
  const int tid  = threadIdx.x;
  const int wave = tid >> 5;
  const int lane = tid & 31;
  const int hi   = lane >> 4;         // half-wave select (ISA 16-bit layouts)
  const int mr   = lane & 15;
  const int bM = blockIdx.x * 64;
  const int bN = blockIdx.y * 16;

  union { v8f v; float f[8]; } c;
#pragma unroll
  for (int i = 0; i < 8; ++i) c.f[i] = 0.0f;

  for (int kt = 0; kt < Kp; kt += 32) {
    const _Float16* gA = Ah  + (size_t)bM * Kp + kt;
    const _Float16* gB = WhT + (size_t)bN * Kp + kt;
#ifdef HAVE_ASYNC_LDS
    // A tile: 64 rows x 64B = 256 x 16B chunks, 2 per thread (async DMA)
#pragma unroll
    for (int i = 0; i < 2; ++i) {
      int q = tid + i * 128;                 // 0..255
      int r = q >> 2, cc = (q & 3) * 8;      // 8 halves = 16B
      __builtin_amdgcn_global_load_async_to_lds_b128(
          (AS1 v4i*)(gA + (size_t)r * Kp + cc),
          (AS3 v4i*)(&Alds[r][cc]), 0, 0);
    }
    // B tile: 16 rows x 64B = 64 x 16B chunks, waves 0-1 (async DMA)
    if (tid < 64) {
      int r = tid >> 2, cc = (tid & 3) * 8;
      __builtin_amdgcn_global_load_async_to_lds_b128(
          (AS1 v4i*)(gB + (size_t)r * Kp + cc),
          (AS3 v4i*)(&Blds[r][cc]), 0, 0);
    }
#if __has_builtin(__builtin_amdgcn_s_wait_asynccnt)
    __builtin_amdgcn_s_wait_asynccnt(0);
#else
    asm volatile("s_wait_asynccnt 0" ::: "memory");
#endif
#else
#pragma unroll
    for (int i = 0; i < 2; ++i) {
      int q = tid + i * 128;
      int r = q >> 2, cc = (q & 3) * 8;
      *(uint4*)&Alds[r][cc] = *(const uint4*)(gA + (size_t)r * Kp + cc);
    }
    if (tid < 64) {
      int r = tid >> 2, cc = (tid & 3) * 8;
      *(uint4*)&Blds[r][cc] = *(const uint4*)(gB + (size_t)r * Kp + cc);
    }
#endif
    __syncthreads();

    // Fragments per CDNA5 ISA 7.12.2 (wave32, 16-bit A 16x32 / B 32x16)
    union { v16h v; _Float16 h[16]; } a, b;
    const int rA = wave * 16 + mr;
#pragma unroll
    for (int j = 0; j < 8; ++j) {
      a.h[j]     = Alds[rA][hi * 8 + j];        // K = hi*8 + j
      a.h[8 + j] = Alds[rA][16 + hi * 8 + j];   // K = 16 + hi*8 + j
    }
#pragma unroll
  for (int j = 0; j < 16; ++j) b.h[j] = Blds[mr][((j >> 3) << 4) + hi * 8 + (j & 7)];

    c.v = __builtin_amdgcn_wmma_f32_16x16x32_f16(false, a.v, false, b.v,
                                                 (short)0, c.v, false, false);
    __syncthreads();
  }

  // D layout: VGPR r -> M = r + 8*hi, N = lane%16
#pragma unroll
  for (int r = 0; r < 8; ++r) {
    int row = bM + wave * 16 + r + 8 * hi;
    int col = bN + mr;
    if (row < M && col < N)
      C[(size_t)row * N + col] = c.f[r] + bias[col];
  }
}

// ---------------------------------------------------------------------------
// Direction permutations for an HxW grid (8 scan orders), computed on-device.
// perms[d*L + pos] = original linear index of the token at permuted pos.
// ---------------------------------------------------------------------------
__global__ void k_perms(int* __restrict__ perms, int H, int W, int L) {
  int idx = blockIdx.x * blockDim.x + threadIdx.x;
  if (idx >= L) return;
  int i = idx / W, j = idx % W;

  int pos_tb = j * H + i;  // transpose-ravel

  // diagonal orders: sort by (key, i); key1 = i+j, key2 = i + (W-1-j)
  int d1 = i + j;
  int d2 = i + (W - 1 - j);
  int S1 = 0, S2 = 0;
  int dmax = (d1 > d2) ? d1 : d2;
  for (int e = 0; e < dmax; ++e) {
    int lo = (e - (W - 1) > 0) ? e - (W - 1) : 0;
    int hi = (e < H - 1) ? e : H - 1;
    int cnt = hi - lo + 1;
    if (e < d1) S1 += cnt;
    if (e < d2) S2 += cnt;
  }
  int lo1 = (d1 - (W - 1) > 0) ? d1 - (W - 1) : 0;
  int lo2 = (d2 - (W - 1) > 0) ? d2 - (W - 1) : 0;
  int pos_tlbr = S1 + (i - lo1);
  int pos_trbl = S2 + (i - lo2);

  perms[0 * L + idx]              = idx;  // lr
  perms[1 * L + (L - 1 - idx)]    = idx;  // lr reversed
  perms[2 * L + pos_tb]           = idx;  // tb
  perms[3 * L + (L - 1 - pos_tb)] = idx;
  perms[4 * L + pos_tlbr]           = idx; // tlbr
  perms[5 * L + (L - 1 - pos_tlbr)] = idx;
  perms[6 * L + pos_trbl]           = idx; // trbl
  perms[7 * L + (L - 1 - pos_trbl)] = idx;
}

// ---------------------------------------------------------------------------
// Fold dtproj into xproj:  Wcat[128,160] = [xproj[:, :4]@dtproj_w | Bc | Cc],
// bcat[160] = [dtproj_b | 0].
// ---------------------------------------------------------------------------
__global__ void k_prep(const float* __restrict__ xproj_w,
                       const float* __restrict__ dtproj_w,
                       const float* __restrict__ dtproj_b,
                       float* __restrict__ Wcat, float* __restrict__ bcat) {
  int idx = blockIdx.x * blockDim.x + threadIdx.x;
  if (idx >= 128 * 160) return;
  int k = idx / 160, n = idx % 160;
  float v;
  if (n < 128) {
    v = 0.0f;
#pragma unroll
    for (int r = 0; r < 4; ++r)
      v += xproj_w[k * 36 + r] * dtproj_w[r * 128 + n];
  } else if (n < 144) {
    v = xproj_w[k * 36 + 4 + (n - 128)];   // Bc columns
  } else {
    v = xproj_w[k * 36 + 20 + (n - 144)];  // Cc columns
  }
  Wcat[idx] = v;
  if (k == 0) bcat[n] = (n < 128) ? dtproj_b[n] : 0.0f;
}

// im2col stage 1: x(2,3,224,224), k=7,s=4,p=2 -> t(B,3136,147)
__global__ void k_unfold1(const float* __restrict__ x, float* __restrict__ t,
                          int total) {
  int idx = blockIdx.x * blockDim.x + threadIdx.x;
  if (idx >= total) return;
  int f  = idx % 147;
  int l  = (idx / 147) % 3136;
  int b  = idx / (147 * 3136);
  int c  = f / 49, r = f % 49;
  int di = r / 7,  dj = r % 7;
  int ho = l / 56, wo = l % 56;
  int ih = ho * 4 + di - 2;
  int iw = wo * 4 + dj - 2;
  float v = 0.0f;
  if (ih >= 0 && ih < 224 && iw >= 0 && iw < 224)
    v = x[((size_t)(b * 3 + c) * 224 + ih) * 224 + iw];
  t[idx] = v;
}

// im2col stages 2/3 (k=3,s=2,p=1) reading token-major f(B, Hp*Wp, 64)
__global__ void k_unfold_t(const float* __restrict__ fprev, float* __restrict__ t,
                           int Hp, int Wp, int Ho, int Wo, int total) {
  int idx = blockIdx.x * blockDim.x + threadIdx.x;
  if (idx >= total) return;
  int L  = Ho * Wo;
  int f  = idx % 576;
  int l  = (idx / 576) % L;
  int b  = idx / (576 * L);
  int c  = f / 9, r = f % 9;
  int di = r / 3, dj = r % 3;
  int ho = l / Wo, wo = l % Wo;
  int ih = ho * 2 + di - 1;
  int iw = wo * 2 + dj - 1;
  float v = 0.0f;
  if (ih >= 0 && ih < Hp && iw >= 0 && iw < Wp)
    v = fprev[((size_t)b * (Hp * Wp) + ih * Wp + iw) * 64 + c];
  t[idx] = v;
}

// LayerNorm over last dim 64, in place. One thread per token.
__global__ void k_layernorm(float* __restrict__ h, const float* __restrict__ g,
                            const float* __restrict__ b, int M) {
  int t = blockIdx.x * blockDim.x + threadIdx.x;
  if (t >= M) return;
  float* row = h + (size_t)t * 64;
  float m = 0.0f;
#pragma unroll 8
  for (int i = 0; i < 64; ++i) m += row[i];
  m *= (1.0f / 64.0f);
  float v = 0.0f;
#pragma unroll 8
  for (int i = 0; i < 64; ++i) { float d = row[i] - m; v += d * d; }
  v *= (1.0f / 64.0f);
  float rs = rsqrtf(v + 1e-5f);
#pragma unroll 8
  for (int i = 0; i < 64; ++i) row[i] = (row[i] - m) * rs * g[i] + b[i];
}

// Depthwise causal conv (K=4) + SiLU along the permuted sequence order.
__global__ void k_conv(const float* __restrict__ xz, const int* __restrict__ perm,
                       const float* __restrict__ cw, const float* __restrict__ cb,
                       float* __restrict__ xc, int L, int total) {
  int idx = blockIdx.x * blockDim.x + threadIdx.x;
  if (idx >= total) return;
  int c = idx % 128;
  int l = (idx / 128) % L;
  int b = idx / (128 * L);
  float acc = cb[c];
#pragma unroll
  for (int k = 0; k < 4; ++k) {
    int j = l + k - 3;
    if (j >= 0)
      acc += cw[c * 4 + k] * xz[((size_t)b * L + perm[j]) * 256 + c];
  }
  xc[idx] = acc / (1.0f + __expf(-acc));  // silu
}

// Selective scan for one direction; state (16 per channel) in registers.
__global__ void __launch_bounds__(128)
k_scan(const float* __restrict__ dbuf, const float* __restrict__ xc,
       const float* __restrict__ xz, const int* __restrict__ perm,
       const float* __restrict__ A_log, const float* __restrict__ Dp_,
       float* __restrict__ acc, int L) {
  const int b = blockIdx.x;
  const int c = threadIdx.x;   // 0..127
  float Arow[16];
#pragma unroll
  for (int s = 0; s < 16; ++s) Arow[s] = -__expf(A_log[c * 16 + s]);
  float hst[16];
#pragma unroll
  for (int s = 0; s < 16; ++s) hst[s] = 0.0f;
  const float Dpv = Dp_[c];

  __shared__ float sB[16], sC[16];
  for (int l = 0; l < L; ++l) {
    const size_t base = (size_t)b * L + l;
    const float* drow = dbuf + base * 160;
    __syncthreads();
    if (c < 16)       sB[c]      = drow[128 + c];
    else if (c < 32)  sC[c - 16] = drow[144 + (c - 16)];
    __syncthreads();

    float dtp = drow[c];
    float dt  = (dtp > 20.0f) ? dtp : __logf(1.0f + __expf(dtp));  // softplus
    float xcv = xc[base * 128 + c];
    float dtx = dt * xcv;
    float y = 0.0f;
#pragma unroll
    for (int s = 0; s < 16; ++s) {
      hst[s] = hst[s] * __expf(dt * Arow[s]) + dtx * sB[s];
      y += hst[s] * sC[s];
    }
    int orig = perm[l];
    float z  = xz[((size_t)b * L + orig) * 256 + 128 + c];
    float sz = z / (1.0f + __expf(-z));
    acc[((size_t)b * L + orig) * 128 + c] += (y + Dpv * xcv) * sz * 0.125f;
  }
}

__global__ void k_zero(float* __restrict__ p, int n) {
  int i = blockIdx.x * blockDim.x + threadIdx.x;
  if (i < n) p[i] = 0.0f;
}

// ---------------------------------------------------------------------------
// Host side
// ---------------------------------------------------------------------------
struct MambaParams {
  const float *lin_w, *lin_b, *ln_g, *ln_b, *inproj_w, *inproj_b,
              *conv_w, *conv_b, *xproj_w, *dtproj_w, *dtproj_b,
              *A_log, *Dp, *outproj_w, *outproj_b;
};

static MambaParams get_params(void* const* d_in, int base) {
  MambaParams p;
  p.lin_w     = (const float*)d_in[base + 0];
  p.lin_b     = (const float*)d_in[base + 1];
  p.ln_g      = (const float*)d_in[base + 2];
  p.ln_b      = (const float*)d_in[base + 3];
  p.inproj_w  = (const float*)d_in[base + 4];
  p.inproj_b  = (const float*)d_in[base + 5];
  p.conv_w    = (const float*)d_in[base + 6];
  p.conv_b    = (const float*)d_in[base + 7];
  p.xproj_w   = (const float*)d_in[base + 8];
  p.dtproj_w  = (const float*)d_in[base + 9];
  p.dtproj_b  = (const float*)d_in[base + 10];
  p.A_log     = (const float*)d_in[base + 11];
  p.Dp        = (const float*)d_in[base + 12];
  p.outproj_w = (const float*)d_in[base + 13];
  p.outproj_b = (const float*)d_in[base + 14];
  return p;
}

// cast+pad A, cast+pad+transpose W, then WMMA GEMM
static void gemm_f16(const float* A, const float* W, const float* bias,
                     float* C, int M, int N, int K,
                     _Float16* Ah, _Float16* WhT, hipStream_t st) {
  int Mp = (M + 63) & ~63;
  int Kp = (K + 31) & ~31;
  int Np = (N + 15) & ~15;
  int tA = Mp * Kp, tW = Np * Kp;
  k_cast_pad <<<dim3((tA + 255) / 256), dim3(256), 0, st>>>(A, Ah, M, K, Kp, tA);
  k_cast_padT<<<dim3((tW + 255) / 256), dim3(256), 0, st>>>(W, WhT, K, N, Kp, tW);
  k_gemm_f16<<<dim3(Mp / 64, Np / 16), dim3(128), 0, st>>>(Ah, WhT, bias, C,
                                                           M, N, Kp);
}

static void run_stage(const float* tok, int B, int L, int Cin, int H, int W,
                      const MambaParams& P, int* perms, float* hbuf,
                      float* xzbuf, float* xcbuf, float* dbuf, float* accbuf,
                      float* Wcat, float* bcat, _Float16* Ah, _Float16* WhT,
                      float* fout, hipStream_t st) {
  const int BL = B * L;
  k_perms<<<dim3((L + 127) / 128), dim3(128), 0, st>>>(perms, H, W, L);
  k_prep<<<dim3(160), dim3(128), 0, st>>>(P.xproj_w, P.dtproj_w, P.dtproj_b,
                                          Wcat, bcat);
  // h = LN(tok @ lin_w + lin_b)   [direction-invariant, computed once]
  gemm_f16(tok, P.lin_w, P.lin_b, hbuf, BL, 64, Cin, Ah, WhT, st);
  k_layernorm<<<dim3((BL + 127) / 128), dim3(128), 0, st>>>(hbuf, P.ln_g,
                                                            P.ln_b, BL);
  // xz = h @ inproj + b  (xi | z), once per stage
  gemm_f16(hbuf, P.inproj_w, P.inproj_b, xzbuf, BL, 256, 64, Ah, WhT, st);

  k_zero<<<dim3((BL * 128 + 255) / 256), dim3(256), 0, st>>>(accbuf, BL * 128);

  for (int d = 0; d < 8; ++d) {
    const int* pm = perms + d * L;
    k_conv<<<dim3((BL * 128 + 255) / 256), dim3(256), 0, st>>>(
        xzbuf, pm, P.conv_w, P.conv_b, xcbuf, L, BL * 128);
    // [dt_pre | Bc | Cc] = xc @ Wcat + bcat
    gemm_f16(xcbuf, Wcat, bcat, dbuf, BL, 160, 128, Ah, WhT, st);
    k_scan<<<dim3(B), dim3(128), 0, st>>>(dbuf, xcbuf, xzbuf, pm, P.A_log,
                                          P.Dp, accbuf, L);
  }
  // mean folded into scan (x0.125); single outproj after accumulation
  gemm_f16(accbuf, P.outproj_w, P.outproj_b, fout, BL, 64, 128, Ah, WhT, st);
}

extern "C" void kernel_launch(void* const* d_in, const int* in_sizes, int n_in,
                              void* d_out, int out_size, void* d_ws,
                              size_t ws_size, hipStream_t stream) {
  (void)in_sizes; (void)n_in; (void)out_size; (void)ws_size;
  const float* x = (const float*)d_in[0];
  MambaParams p1 = get_params(d_in, 1);
  MambaParams p2 = get_params(d_in, 16);
  const float* proj_w = (const float*)d_in[31];
  const float* proj_b = (const float*)d_in[32];
  float* out = (float*)d_out;

  const int B = 2;
  const int H1 = 56, W1 = 56, L1 = H1 * W1;    // 3136
  const int H2 = 28, W2 = 28, L2 = H2 * W2;    // 784
  const int H3 = 14, W3 = 14, L3 = H3 * W3;    // 196

  // workspace carve (stage-1 maxima; buffers reused across stages/dirs)
  char* wb = (char*)d_ws;
  size_t off = 0;
  auto alloc = [&](size_t bytes) -> void* {
    size_t cur = (off + 255) & ~(size_t)255;
    off = cur + bytes;
    return (void*)(wb + cur);
  };
  int*      perms  = (int*)     alloc((size_t)8 * L1 * sizeof(int));
  float*    tbuf   = (float*)   alloc((size_t)B * L1 * 147 * sizeof(float));
  float*    hbuf   = (float*)   alloc((size_t)B * L1 * 64  * sizeof(float));
  float*    xzbuf  = (float*)   alloc((size_t)B * L1 * 256 * sizeof(float));
  float*    xcbuf  = (float*)   alloc((size_t)B * L1 * 128 * sizeof(float));
  float*    dbuf   = (float*)   alloc((size_t)B * L1 * 160 * sizeof(float));
  float*    accbuf = (float*)   alloc((size_t)B * L1 * 128 * sizeof(float));
  float*    f1     = (float*)   alloc((size_t)B * L1 * 64  * sizeof(float));
  float*    f2     = (float*)   alloc((size_t)B * L2 * 64  * sizeof(float));
  float*    Wcat   = (float*)   alloc((size_t)128 * 160 * sizeof(float));
  float*    bcat   = (float*)   alloc((size_t)256 * sizeof(float));
  // f16 GEMM scratch: max Mp*Kp = 6272*160; max Np*Kp = 768*576
  _Float16* Ah     = (_Float16*)alloc((size_t)6272 * 160 * sizeof(_Float16));
  _Float16* WhT    = (_Float16*)alloc((size_t)768 * 576 * sizeof(_Float16));

  // ---- stage 1 ----
  {
    int total = B * L1 * 147;
    k_unfold1<<<dim3((total + 255) / 256), dim3(256), 0, stream>>>(x, tbuf,
                                                                   total);
    run_stage(tbuf, B, L1, 147, H1, W1, p1, perms, hbuf, xzbuf, xcbuf, dbuf,
              accbuf, Wcat, bcat, Ah, WhT, f1, stream);
  }
  // ---- stage 2 ----
  {
    int total = B * L2 * 576;
    k_unfold_t<<<dim3((total + 255) / 256), dim3(256), 0, stream>>>(
        f1, tbuf, H1, W1, H2, W2, total);
    run_stage(tbuf, B, L2, 576, H2, W2, p2, perms, hbuf, xzbuf, xcbuf, dbuf,
              accbuf, Wcat, bcat, Ah, WhT, f2, stream);
  }
  // ---- stage 3: unfold + final projection ----
  {
    int total = B * L3 * 576;
    k_unfold_t<<<dim3((total + 255) / 256), dim3(256), 0, stream>>>(
        f2, tbuf, H2, W2, H3, W3, total);
    gemm_f16(tbuf, proj_w, proj_b, out, B * L3, 768, 576, Ah, WhT, stream);
  }
}
